// CQ_20100446945509
// MI455X (gfx1250) — compile-verified, matrix operand order checked by reading, not run on hardware
//
#include <hip/hip_runtime.h>
#include <hip/hip_bf16.h>
#include <stdint.h>

// ---------------------------------------------------------------------------
// CDNA5 (gfx1250) implementation of the residual-quantizer transformer.
// Heavy compute = bf16 WMMA GEMMs (v_wmma_f32_16x16x32_bf16), wave32.
// B operand staged through LDS (conflict-free padded tile), A streamed.
// ---------------------------------------------------------------------------

typedef __attribute__((ext_vector_type(16))) __bf16 v16bf;
typedef __attribute__((ext_vector_type(8)))  float  v8f;

#define NSAMP   32768      // B*H*W = 32*32*32
#define DMODEL  256
#define VOCABN  4096
#define TOKMAX  (NSAMP*4)

#define KCHUNK  256
#define BPITCH  (KCHUNK + 8)        // bf16 elements; 528B row pitch -> 4-bank stagger

// ---------------------------------------------------------------------------
// WMMA GEMM:  C[M, Nc] = A[M, K] (bf16, lda) @ W[Nc, K]^T (bf16, ldw=K)
// Block: 8 waves, 128 M-rows x 64 N-cols.  W tile staged in LDS per block.
// Fragment layouts follow CDNA5 ISA 7.12.2 (16-bit A 16x32; B mirrored).
// Per k-step: load A frag + all 4 B frags, then 4 back-to-back WMMAs.
// ---------------------------------------------------------------------------
__global__ __launch_bounds__(256) void k_gemm(
    const __bf16* __restrict__ A, int lda,
    const __bf16* __restrict__ W, int K,
    const float*  __restrict__ bias,
    float*  __restrict__ outF,
    __bf16* __restrict__ outB,
    int ldc, int cc0,
    int M, int Nc, int relu)
{
    __shared__ __bf16 ws[64 * BPITCH];          // 33,792 bytes

    const int tid    = threadIdx.x;
    const int lane   = tid & 31;
    const int wave   = tid >> 5;
    const int tilesMg = (M >> 4) >> 3;          // groups of 8 M-tiles (128 rows)
    const int ni = blockIdx.x / tilesMg;
    const int mg = blockIdx.x % tilesMg;
    const int mi = mg * 8 + wave;               // this wave's 16-row M tile
    const int ncol0 = ni * 64;

    const int mrowA = mi * 16 + (lane & 15);
    const int khalf = (lane >> 4) * 8;          // lanes 16-31 hold K+8 group

    union Frag { v16bf v; unsigned u[8]; };

    v8f acc0 = {}, acc1 = {}, acc2 = {}, acc3 = {};

    for (int kc = 0; kc < K; kc += KCHUNK) {
        __syncthreads();
        // Cooperative load of W[ncol0..+64][kc..kc+KCHUNK) into padded LDS tile.
        {
            const int chunksPerRow = KCHUNK / 8;        // 16B chunks (8 bf16)
            const int total = 64 * chunksPerRow;        // 2048
            for (int c = tid; c < total; c += 256) {
                const int row = c / chunksPerRow;
                const int ck  = c % chunksPerRow;
                const uint4 v = *(const uint4*)(W + (size_t)(ncol0 + row) * K + kc + ck * 8);
                *(uint4*)(&ws[row * BPITCH + ck * 8]) = v;
            }
        }
        __syncthreads();

        if (kc + KCHUNK < K)   // prefetch next A chunk (global_prefetch_b8)
            __builtin_prefetch(A + (size_t)mrowA * lda + kc + KCHUNK, 0, 0);

        for (int kk = 0; kk < KCHUNK; kk += 32) {
            // ---- gather all operands first (one wait, then 4 WMMAs) ----
            Frag a;
            const __bf16* ap = A + (size_t)mrowA * lda + kc + kk + khalf;
            #pragma unroll
            for (int j = 0; j < 8; ++j) {
                const int k0 = (j < 4) ? (2 * j) : (16 + 2 * (j - 4));
                a.u[j] = *(const unsigned*)(ap + k0);
            }
            Frag b0, b1, b2, b3;
            {
                const __bf16* wp0 = &ws[((lane & 15) +  0) * BPITCH + kk + khalf];
                const __bf16* wp1 = &ws[((lane & 15) + 16) * BPITCH + kk + khalf];
                const __bf16* wp2 = &ws[((lane & 15) + 32) * BPITCH + kk + khalf];
                const __bf16* wp3 = &ws[((lane & 15) + 48) * BPITCH + kk + khalf];
                #pragma unroll
                for (int j = 0; j < 8; ++j) {
                    const int k0 = (j < 4) ? (2 * j) : (16 + 2 * (j - 4));
                    b0.u[j] = *(const unsigned*)(wp0 + k0);
                    b1.u[j] = *(const unsigned*)(wp1 + k0);
                    b2.u[j] = *(const unsigned*)(wp2 + k0);
                    b3.u[j] = *(const unsigned*)(wp3 + k0);
                }
            }
            acc0 = __builtin_amdgcn_wmma_f32_16x16x32_bf16(
                       false, a.v, false, b0.v, (short)0, acc0, false, false);
            acc1 = __builtin_amdgcn_wmma_f32_16x16x32_bf16(
                       false, a.v, false, b1.v, (short)0, acc1, false, false);
            acc2 = __builtin_amdgcn_wmma_f32_16x16x32_bf16(
                       false, a.v, false, b2.v, (short)0, acc2, false, false);
            acc3 = __builtin_amdgcn_wmma_f32_16x16x32_bf16(
                       false, a.v, false, b3.v, (short)0, acc3, false, false);
        }
    }

    // Epilogue.  C layout: lanes0-15 VGPR r -> M=r ; lanes16-31 -> M=8+r ; N=lane&15.
    const int mbase = mi * 16 + ((lane >> 4) ? 8 : 0);
    #pragma unroll
    for (int t = 0; t < 4; ++t) {
        const int n  = ni * 64 + t * 16 + (lane & 15);
        const float bv = bias ? bias[n] : 0.0f;
        v8f c = (t == 0) ? acc0 : (t == 1) ? acc1 : (t == 2) ? acc2 : acc3;
        #pragma unroll
        for (int r = 0; r < 8; ++r) {
            float v = c[r] + bv;
            if (relu) v = fmaxf(v, 0.0f);
            const size_t off = (size_t)(mbase + r) * ldc + cc0 + n;
            if (outF) outF[off] = v;
            if (outB) outB[off] = (__bf16)v;
        }
    }
}

// ---------------------------------------------------------------------------
// Small helper kernels
// ---------------------------------------------------------------------------
__global__ void k_cvt(const float* __restrict__ s, __bf16* __restrict__ d, int n) {
    int i = blockIdx.x * blockDim.x + threadIdx.x;
    if (i < n) d[i] = (__bf16)s[i];
}
__global__ void k_copyf(const float* __restrict__ s, float* __restrict__ d, int n) {
    int i = blockIdx.x * blockDim.x + threadIdx.x;
    if (i < n) d[i] = s[i];
}
__global__ void k_init_out(float* out) {
    out[1048576] = 0.0f;   // vq_loss
    out[1048577] = 0.0f;   // uniform_loss
}

// tanh input projection: x (B,Cz,H,W) -> xt (N,4)
__global__ void k_xt(const float* __restrict__ x, const float* __restrict__ pw,
                     const float* __restrict__ pb, float* __restrict__ xt,
                     float* __restrict__ codef, int* __restrict__ gid) {
    int n = blockIdx.x * blockDim.x + threadIdx.x;
    if (n >= NSAMP) return;
    int b = n >> 10, l = n & 1023;
    float xin[32];
    #pragma unroll
    for (int c = 0; c < 32; ++c) xin[c] = x[(size_t)b * 32768 + c * 1024 + l];
    #pragma unroll
    for (int j = 0; j < 4; ++j) {
        float a = pb[j];
        #pragma unroll
        for (int c = 0; c < 32; ++c) a += xin[c] * pw[j * 32 + c];
        xt[n * 4 + j] = tanhf(a);
    }
    #pragma unroll
    for (int j = 0; j < 4; ++j) codef[n * 4 + j] = 0.0f;
    gid[n] = 0;
}

// h = seq @ in_w.T + in_b   (K=2, elementwise)
__global__ void k_hinit(const float* __restrict__ codef, const float* __restrict__ xt,
                        const float* __restrict__ inw, const float* __restrict__ inb,
                        float* __restrict__ h, int M, int S) {
    int idx = blockIdx.x * blockDim.x + threadIdx.x;
    if (idx >= M * DMODEL) return;
    int m = idx >> 8, d = idx & 255;
    int n = m / S, t = m - n * S;
    float c0 = codef[n * 4 + t], x1 = xt[n * 4 + t];
    h[(size_t)m * DMODEL + d] = c0 * inw[d * 2] + x1 * inw[d * 2 + 1] + inb[d];
}

// RoPE: writes rot (bf16) and current h (bf16)
__global__ void k_rope(const float* __restrict__ h, __bf16* __restrict__ rot,
                       __bf16* __restrict__ hbf, int M, int S) {
    int idx = blockIdx.x * blockDim.x + threadIdx.x;
    if (idx >= M * 128) return;
    int m = idx >> 7, j = idx & 127;
    int t = m % S;
    float invf = __expf(-(float)(2 * j) * (1.0f / 256.0f) * 9.210340371976184f);
    float ang  = (float)t * invf;
    float sn, cs;
    __sincosf(ang, &sn, &cs);
    size_t base = (size_t)m * DMODEL + 2 * j;
    float h0 = h[base], h1 = h[base + 1];
    rot[base]     = (__bf16)(h0 * cs - h1 * sn);
    rot[base + 1] = (__bf16)(h1 * cs + h0 * sn);
    hbf[base]     = (__bf16)h0;
    hbf[base + 1] = (__bf16)h1;
}

// Causal attention per (sample, head); S <= 4 so this is trivial VALU work.
__global__ void k_attn(const __bf16* __restrict__ qkv, __bf16* __restrict__ obf, int S) {
    int idx = blockIdx.x * blockDim.x + threadIdx.x;
    if (idx >= NSAMP * 8) return;
    int n = idx >> 3, hh = idx & 7;
    const float scale = 0.17677669529663687f;   // 1/sqrt(32)
    for (int s = 0; s < S; ++s) {
        const __bf16* qp = qkv + (size_t)(n * S + s) * 768 + hh * 32;
        float q[32];
        #pragma unroll
        for (int d = 0; d < 32; ++d) q[d] = (float)qp[d];
        float sc[4]; float mx = -1e30f;
        for (int t = 0; t <= s; ++t) {
            const __bf16* kp = qkv + (size_t)(n * S + t) * 768 + 256 + hh * 32;
            float dot = 0.0f;
            #pragma unroll
            for (int d = 0; d < 32; ++d) dot += q[d] * (float)kp[d];
            sc[t] = dot * scale;
            mx = fmaxf(mx, sc[t]);
        }
        float den = 0.0f;
        for (int t = 0; t <= s; ++t) { sc[t] = __expf(sc[t] - mx); den += sc[t]; }
        float o[32];
        #pragma unroll
        for (int d = 0; d < 32; ++d) o[d] = 0.0f;
        for (int t = 0; t <= s; ++t) {
            const __bf16* vp = qkv + (size_t)(n * S + t) * 768 + 512 + hh * 32;
            float wgt = sc[t] / den;
            #pragma unroll
            for (int d = 0; d < 32; ++d) o[d] += wgt * (float)vp[d];
        }
        __bf16* op = obf + (size_t)(n * S + s) * DMODEL + hh * 32;
        #pragma unroll
        for (int d = 0; d < 32; ++d) op[d] = (__bf16)o[d];
    }
}

// LayerNorm(h + add): one wave32 per token, shuffle reduction.
__global__ __launch_bounds__(256) void k_ln(float* __restrict__ h,
                                            const float* __restrict__ add,
                                            __bf16* __restrict__ hbf,
                                            const float* __restrict__ g,
                                            const float* __restrict__ b, int M) {
    int wave = (blockIdx.x * 256 + threadIdx.x) >> 5;
    int lane = threadIdx.x & 31;
    if (wave >= M) return;
    size_t base = (size_t)wave * DMODEL + lane * 8;
    float x[8]; float s = 0.0f;
    #pragma unroll
    for (int j = 0; j < 8; ++j) { x[j] = h[base + j] + add[base + j]; s += x[j]; }
    #pragma unroll
    for (int msk = 16; msk >= 1; msk >>= 1) s += __shfl_xor(s, msk, 32);
    float mu = s * (1.0f / 256.0f);
    float v = 0.0f;
    #pragma unroll
    for (int j = 0; j < 8; ++j) { float d = x[j] - mu; v += d * d; }
    #pragma unroll
    for (int msk = 16; msk >= 1; msk >>= 1) v += __shfl_xor(v, msk, 32);
    v *= (1.0f / 256.0f);
    float inv = rsqrtf(v + 1e-5f);
    #pragma unroll
    for (int j = 0; j < 8; ++j) {
        int d = lane * 8 + j;
        float y = (x[j] - mu) * inv * g[d] + b[d];
        h[base + j] = y;
        hbf[base + j] = (__bf16)y;
    }
}

// out = h @ out_w.T + out_b, only last position, 9 cols.
__global__ void k_outproj(const float* __restrict__ h, const float* __restrict__ ow,
                          const float* __restrict__ ob, float* __restrict__ zall,
                          float* __restrict__ bnd, int S, int lev) {
    int idx = blockIdx.x * blockDim.x + threadIdx.x;
    if (idx >= NSAMP * 9) return;
    int n = idx / 9, j = idx % 9;
    const float* hp = h + (size_t)(n * S + S - 1) * DMODEL;
    const float* wp = ow + j * DMODEL;
    float acc = 0.0f;
    for (int d = 0; d < DMODEL; ++d) acc += hp[d] * wp[d];
    acc += ob[j];
    if (j == 0) zall[n * 4 + lev] = acc;
    else        bnd[n * 7 + (j - 1)] = acc;
}

__device__ __forceinline__ unsigned encf(float f) {
    unsigned u = __float_as_uint(f);
    return (u >> 31) ? ~u : (u | 0x80000000u);
}
__device__ __forceinline__ float decf(unsigned e) {
    return (e & 0x80000000u) ? __uint_as_float(e ^ 0x80000000u) : __uint_as_float(~e);
}

__global__ void k_initlevel(float* hc, float* hm1, unsigned* mm) {
    int i = blockIdx.x * blockDim.x + threadIdx.x;
    if (i < VOCABN * 8) hc[i] = 0.0f;
    if (i < VOCABN)     hm1[i] = 0.0f;
    if (i == 0) { mm[0] = 0xFFFFFFFFu; mm[1] = 0u; }
}

__global__ void k_minmax(const float* __restrict__ zall, unsigned* mm, int lev) {
    int n = blockIdx.x * blockDim.x + threadIdx.x;
    if (n >= NSAMP) return;
    unsigned e = encf(zall[n * 4 + lev]);
    unsigned lo = e, hi = e;
    #pragma unroll
    for (int msk = 16; msk >= 1; msk >>= 1) {
        unsigned ol = (unsigned)__shfl_xor((int)lo, msk, 32);
        unsigned oh = (unsigned)__shfl_xor((int)hi, msk, 32);
        lo = lo < ol ? lo : ol;
        hi = hi > oh ? hi : oh;
    }
    if ((threadIdx.x & 31) == 0) { atomicMin(&mm[0], lo); atomicMax(&mm[1], hi); }
}

__global__ void k_bins(const float* __restrict__ zall, const float* __restrict__ bnd,
                       const unsigned* __restrict__ mm, const int* __restrict__ gid,
                       int* __restrict__ bin, float* __restrict__ qall,
                       float* __restrict__ codef, float* __restrict__ hc,
                       float* __restrict__ hm1, int lev) {
    int n = blockIdx.x * blockDim.x + threadIdx.x;
    if (n >= NSAMP) return;
    float z  = zall[n * 4 + lev];
    float MN = decf(mm[0]), MX = decf(mm[1]);
    float l[8], r[8];
    l[0] = MN - 0.01f; r[7] = MX + 0.01f;
    #pragma unroll
    for (int j = 0; j < 7; ++j) { float bb = bnd[n * 7 + j]; l[j + 1] = bb; r[j] = bb; }
    int match = -1; float qv = 0.0f;
    #pragma unroll
    for (int j = 0; j < 8; ++j) {
        bool m = (z >= l[j]) && (z < r[j]) && (r[j] > l[j]);
        if (m && match < 0) { match = j; qv = 0.5f * (l[j] + r[j]); }
    }
    bin[n] = match;
    qall[n * 4 + lev] = (match >= 0) ? qv : 0.0f;
    if (lev + 1 < 4) codef[n * 4 + lev + 1] = (float)match;
    int g = gid[n];
    if (g >= 0 && g < VOCABN) {
        if (match >= 0) atomicAdd(&hc[g * 8 + match], 1.0f);
        else            atomicAdd(&hm1[g], 1.0f);
    }
}

__global__ void k_ratios(float* __restrict__ rat, const float* __restrict__ hc,
                         const float* __restrict__ hm1) {
    int v = blockIdx.x * blockDim.x + threadIdx.x;
    if (v >= VOCABN) return;
    float r7 = rat[v * 8 + 7];
    float h7 = hc[v * 8 + 7], m1 = hm1[v];
    float last = (h7 > 0.0f) ? h7 : ((m1 > 0.0f) ? m1 : r7);
    #pragma unroll
    for (int j = 0; j < 8; ++j) { float h = hc[v * 8 + j]; if (h > 0.0f) rat[v * 8 + j] = h; }
    rat[v * 8 + 7] = last;
}

__global__ void k_loss(const float* __restrict__ rat, const int* __restrict__ gid,
                       const float* __restrict__ bnd, float* __restrict__ uloss) {
    int n = blockIdx.x * blockDim.x + threadIdx.x;
    if (n >= NSAMP) return;
    int g = gid[n]; g = g < 0 ? 0 : (g > VOCABN - 1 ? VOCABN - 1 : g);
    float r[8];
    #pragma unroll
    for (int j = 0; j < 8; ++j) r[j] = rat[g * 8 + j];
    float gr[7]; float nrm = 0.0f;
    #pragma unroll
    for (int j = 0; j < 7; ++j) { gr[j] = -(r[j + 1] - r[j]); nrm += gr[j] * gr[j]; }
    float inv = rsqrtf(nrm);
    float dot = 0.0f; float b[7];
    #pragma unroll
    for (int j = 0; j < 7; ++j) { b[j] = bnd[n * 7 + j]; dot += b[j] * gr[j] * inv; }
    float ord = 0.0f;
    #pragma unroll
    for (int j = 0; j < 6; ++j) {
        float d = b[j + 1] - b[j];
        d = fminf(d, 0.1f); d = fmaxf(d, -9999999.0f);
        ord += d;
    }
    atomicAdd(uloss, dot * (1.0f / NSAMP) - ord * (1.0f / (NSAMP * 6.0f)));
}

__global__ void k_gidupd(int* __restrict__ gid, const int* __restrict__ bin, int p8) {
    int n = blockIdx.x * blockDim.x + threadIdx.x;
    if (n >= NSAMP) return;
    gid[n] += bin[n] * p8;
}

// Final: out = q @ proj_post_w.T + b, transposed to (B,Cz,H,W); vq loss.
__global__ void k_final(const float* __restrict__ qall, const float* __restrict__ zall,
                        const float* __restrict__ ppw, const float* __restrict__ ppb,
                        float* __restrict__ out, float* __restrict__ vq) {
    int n = blockIdx.x * blockDim.x + threadIdx.x;
    if (n >= NSAMP) return;
    float q[4], e = 0.0f;
    #pragma unroll
    for (int j = 0; j < 4; ++j) {
        q[j] = qall[n * 4 + j];
        float d = q[j] - zall[n * 4 + j];
        e += d * d;
    }
    atomicAdd(vq, e * (1.0f / (NSAMP * 4.0f)));
    int b = n >> 10, l = n & 1023;
    #pragma unroll
    for (int c = 0; c < 32; ++c) {
        float acc = ppb[c];
        #pragma unroll
        for (int j = 0; j < 4; ++j) acc += q[j] * ppw[c * 4 + j];
        out[(size_t)b * 32768 + c * 1024 + l] = acc;
    }
}

// ---------------------------------------------------------------------------
// Host orchestration
// ---------------------------------------------------------------------------
static inline void gemm(const __bf16* A, int lda, const __bf16* W, int K, int Nc,
                        const float* bias, float* outF, __bf16* outB, int ldc, int cc0,
                        int M, int relu, hipStream_t stream) {
    int blocks = (M / 128) * (Nc / 64);
    k_gemm<<<blocks, 256, 0, stream>>>(A, lda, W, K, bias, outF, outB, ldc, cc0, M, Nc, relu);
}

#define LAUNCH(k, n, ...) k<<<((n) + 255) / 256, 256, 0, stream>>>(__VA_ARGS__)

extern "C" void kernel_launch(void* const* d_in, const int* in_sizes, int n_in,
                              void* d_out, int out_size, void* d_ws, size_t ws_size,
                              hipStream_t stream) {
    (void)in_sizes; (void)n_in; (void)out_size; (void)ws_size;
    const float* x    = (const float*)d_in[0];
    const float* ppw  = (const float*)d_in[1];
    const float* ppb  = (const float*)d_in[2];
    const float* postw= (const float*)d_in[3];
    const float* postb= (const float*)d_in[4];
    const float* inw  = (const float*)d_in[5];
    const float* inb  = (const float*)d_in[6];
    const float* outw = (const float*)d_in[7];
    const float* outb = (const float*)d_in[8];
    const float* qkvw = (const float*)d_in[9];
    const float* qkvb = (const float*)d_in[10];
    const float* aow  = (const float*)d_in[11];
    const float* aob  = (const float*)d_in[12];
    const float* f1w  = (const float*)d_in[13];
    const float* f1b  = (const float*)d_in[14];
    const float* f2w  = (const float*)d_in[15];
    const float* f2b  = (const float*)d_in[16];
    const float* ln1g = (const float*)d_in[17];
    const float* ln1b = (const float*)d_in[18];
    const float* ln2g = (const float*)d_in[19];
    const float* ln2b = (const float*)d_in[20];
    const float* rin  = (const float*)d_in[21];
    float* out = (float*)d_out;

    char*  w   = (char*)d_ws;
    size_t off = 0;
    auto alloc = [&](size_t bytes) -> char* {
        char* p = w + off;
        off += (bytes + 255) & ~(size_t)255;
        return p;
    };
    float*    xtb   = (float*)   alloc((size_t)NSAMP * 4 * 4);
    float*    codef = (float*)   alloc((size_t)NSAMP * 4 * 4);
    int*      gid   = (int*)     alloc((size_t)NSAMP * 4);
    float*    zall  = (float*)   alloc((size_t)NSAMP * 4 * 4);
    float*    qall  = (float*)   alloc((size_t)NSAMP * 4 * 4);
    float*    bnd   = (float*)   alloc((size_t)NSAMP * 7 * 4);
    int*      bin   = (int*)     alloc((size_t)NSAMP * 4);
    float*    hc    = (float*)   alloc((size_t)VOCABN * 8 * 4);
    float*    hm1   = (float*)   alloc((size_t)VOCABN * 4);
    float*    rat   = (float*)   alloc((size_t)VOCABN * 8 * 4);
    unsigned* mm    = (unsigned*)alloc(256);
    __bf16*   qkvwB = (__bf16*)  alloc((size_t)3 * 768 * 256 * 2);
    __bf16*   aowB  = (__bf16*)  alloc((size_t)3 * 256 * 256 * 2);
    __bf16*   f1wB  = (__bf16*)  alloc((size_t)3 * 512 * 256 * 2);
    __bf16*   f2wB  = (__bf16*)  alloc((size_t)3 * 256 * 512 * 2);
    float*    hbuf  = (float*)   alloc((size_t)TOKMAX * 256 * 4);
    __bf16*   hbf   = (__bf16*)  alloc((size_t)TOKMAX * 256 * 2);
    __bf16*   rotbf = (__bf16*)  alloc((size_t)TOKMAX * 256 * 2);
    __bf16*   qkvbf = (__bf16*)  alloc((size_t)TOKMAX * 768 * 2);
    __bf16*   obf   = (__bf16*)  alloc((size_t)TOKMAX * 256 * 2);
    float*    tmpf  = (float*)   alloc((size_t)TOKMAX * 256 * 4);
    __bf16*   t1bf  = (__bf16*)  alloc((size_t)TOKMAX * 512 * 2);

    // One-time prep
    LAUNCH(k_cvt, 3 * 768 * 256, qkvw, qkvwB, 3 * 768 * 256);
    LAUNCH(k_cvt, 3 * 256 * 256, aow,  aowB,  3 * 256 * 256);
    LAUNCH(k_cvt, 3 * 512 * 256, f1w,  f1wB,  3 * 512 * 256);
    LAUNCH(k_cvt, 3 * 256 * 512, f2w,  f2wB,  3 * 256 * 512);
    LAUNCH(k_copyf, VOCABN * 8, rin, rat, VOCABN * 8);
    k_init_out<<<1, 1, 0, stream>>>(out);
    LAUNCH(k_xt, NSAMP, x, ppw, ppb, xtb, codef, gid);

    for (int lev = 0; lev < 4; ++lev) {
        const int S = lev + 1;
        const int M = NSAMP * S;

        LAUNCH(k_hinit, M * 256, codef, xtb, inw, inb, hbuf, M, S);

        for (int l = 0; l < 3; ++l) {
            LAUNCH(k_rope, M * 128, hbuf, rotbf, hbf, M, S);
            // Q,K from rot:  W rows [0,512) of qkv_w[l]
            gemm(rotbf, 256, qkvwB + (size_t)l * 768 * 256, 256, 512,
                 qkvb + l * 768, nullptr, qkvbf, 768, 0, M, 0, stream);
            // V from h:      W rows [512,768)
            gemm(hbf, 256, qkvwB + (size_t)l * 768 * 256 + 512 * 256, 256, 256,
                 qkvb + l * 768 + 512, nullptr, qkvbf, 768, 512, M, 0, stream);
            LAUNCH(k_attn, NSAMP * 8, qkvbf, obf, S);
            gemm(obf, 256, aowB + (size_t)l * 256 * 256, 256, 256,
                 aob + l * 256, tmpf, nullptr, 256, 0, M, 0, stream);
            k_ln<<<M / 8, 256, 0, stream>>>(hbuf, tmpf, hbf, ln1g + l * 256, ln1b + l * 256, M);
            gemm(hbf, 256, f1wB + (size_t)l * 512 * 256, 256, 512,
                 f1b + l * 512, nullptr, t1bf, 512, 0, M, 1, stream);
            gemm(t1bf, 512, f2wB + (size_t)l * 256 * 512, 512, 256,
                 f2b + l * 256, tmpf, nullptr, 256, 0, M, 0, stream);
            k_ln<<<M / 8, 256, 0, stream>>>(hbuf, tmpf, hbf, ln2g + l * 256, ln2b + l * 256, M);
        }

        LAUNCH(k_outproj, NSAMP * 9, hbuf, outw, outb, zall, bnd, S, lev);
        LAUNCH(k_initlevel, VOCABN * 8, hc, hm1, mm);
        LAUNCH(k_minmax, NSAMP, zall, mm, lev);
        LAUNCH(k_bins, NSAMP, zall, bnd, mm, gid, bin, qall, codef, hc, hm1, lev);
        LAUNCH(k_ratios, VOCABN, rat, hc, hm1);
        LAUNCH(k_loss, NSAMP, rat, gid, bnd, out + 1048577);
        const int p8 = 1 << (3 * lev);
        LAUNCH(k_gidupd, NSAMP, gid, bin, p8);
    }

    LAUNCH(k_final, NSAMP, qall, zall, postw, postb, out, out + 1048576);
}
